// SlidingRNN_v1_27676769255799
// MI455X (gfx1250) — matrix operand order repaired and checked by reading, not compiled, not run on hardware
//
#include <hip/hip_runtime.h>
#include <hip/hip_bf16.h>

// Problem constants (match reference)
#define BB 128   // batch
#define TT 64    // timesteps
#define CC 66    // input channels
#define SS 512   // hidden size
#define WW 32    // window
#define CP 96    // padded channel count (multiple of 32 for WMMA K)

typedef __attribute__((ext_vector_type(16))) __bf16 v16bf;
typedef __attribute__((ext_vector_type(8)))  __bf16 v8bf;
typedef __attribute__((ext_vector_type(8)))  float  v8f;

// ---------------------------------------------------------------------------
// WMMA fragment loaders (contiguous k order k = 16*(lane/16) + s for A and B; not the ISA table order)
// A (16x32, MxK): lanes 0-15 -> M=lane, K {0..15};
//                 lanes 16-31 -> same M, K {16..31} (matches B's K order).
// => per lane: two contiguous 8-element (16B) loads at kbase and kbase+8.
// ---------------------------------------------------------------------------
static __device__ __forceinline__ v16bf load_a_frag(const __bf16* __restrict__ base,
                                                    int stride, int r0, int k0, int lane) {
  const int m  = lane & 15;
  const int kb = (lane >> 4) << 4;
  const __bf16* p = base + (size_t)(r0 + m) * stride + k0 + kb;
  v8bf lo = *(const v8bf*)(p);
  v8bf hi = *(const v8bf*)(p + 8);
  v16bf a;
#pragma unroll
  for (int i = 0; i < 8; ++i) { a[i] = lo[i]; a[i + 8] = hi[i]; }
  return a;
}

// B (32x16, KxN): lane n = lane&15 selects column (weight row), lanes 0-15 hold
// K=0..15, lanes 16-31 hold K=16..31 => 16 contiguous bf16 per lane (32B load).
// We compute out[b,n] = sum_k in[b,k] * W[n,k], so B[k][n] = W[row0+n][k].
static __device__ __forceinline__ v16bf load_b_frag(const __bf16* __restrict__ W,
                                                    int Kp, int row0, int k0, int lane) {
  const int n  = lane & 15;
  const int kh = (lane >> 4) << 4;
  const __bf16* p = W + (size_t)(row0 + n) * Kp + k0 + kh;
  return *(const v16bf*)(p);
}

static __device__ __forceinline__ float fsigmoid(float x) {
  return 1.0f / (1.0f + __expf(-x));
}

// ---------------------------------------------------------------------------
// Fused GRU cell: one block owns a 16-wide hidden-column tile (grid = S/16 = 32),
// 8 waves each own a 16-row batch tile.  Each wave accumulates 6 gate tiles
// (x@Wih.T and h@Whh.T for r/z/n) with v_wmma_f32_16x16x32_bf16, then applies
// the gate nonlinearities in-register and writes h_new (f32 + bf16 copies).
// ---------------------------------------------------------------------------
__launch_bounds__(256)
__global__ void gru_cell_wmma(const __bf16* __restrict__ inA, int strideA, int Kin,
                              const __bf16* __restrict__ hB,   // h_prev bf16 (B x S)
                              const float*  __restrict__ hF,   // h_prev f32  (B x S)
                              const __bf16* __restrict__ Wih,  // (3S x Kin) bf16
                              const __bf16* __restrict__ Whh,  // (3S x S)   bf16
                              const float*  __restrict__ bih,
                              const float*  __restrict__ bhh,
                              float*  __restrict__ outF,
                              __bf16* __restrict__ outB,
                              float*  __restrict__ extra)      // optional ring slot
{
  const int lane = threadIdx.x & 31;
  const int wave = threadIdx.x >> 5;
  const int r0   = wave * 16;          // batch row tile
  const int col0 = blockIdx.x * 16;    // hidden column tile

  v8f air = {}, aiz = {}, ain = {}, ahr = {}, ahz = {}, ahn = {};

  // gi = x @ Wih.T  (three gate row-blocks of Wih share the A fragment)
  for (int k0 = 0; k0 < Kin; k0 += 32) {
    v16bf a  = load_a_frag(inA, strideA, r0, k0, lane);
    v16bf br = load_b_frag(Wih, Kin, 0 * SS + col0, k0, lane);
    v16bf bz = load_b_frag(Wih, Kin, 1 * SS + col0, k0, lane);
    v16bf bn = load_b_frag(Wih, Kin, 2 * SS + col0, k0, lane);
    air = __builtin_amdgcn_wmma_f32_16x16x32_bf16(false, a, false, br, (short)0, air, false, false);
    aiz = __builtin_amdgcn_wmma_f32_16x16x32_bf16(false, a, false, bz, (short)0, aiz, false, false);
    ain = __builtin_amdgcn_wmma_f32_16x16x32_bf16(false, a, false, bn, (short)0, ain, false, false);
  }
  // gh = h @ Whh.T
  for (int k0 = 0; k0 < SS; k0 += 32) {
    v16bf a  = load_a_frag(hB, SS, r0, k0, lane);
    v16bf br = load_b_frag(Whh, SS, 0 * SS + col0, k0, lane);
    v16bf bz = load_b_frag(Whh, SS, 1 * SS + col0, k0, lane);
    v16bf bn = load_b_frag(Whh, SS, 2 * SS + col0, k0, lane);
    ahr = __builtin_amdgcn_wmma_f32_16x16x32_bf16(false, a, false, br, (short)0, ahr, false, false);
    ahz = __builtin_amdgcn_wmma_f32_16x16x32_bf16(false, a, false, bz, (short)0, ahz, false, false);
    ahn = __builtin_amdgcn_wmma_f32_16x16x32_bf16(false, a, false, bn, (short)0, ahn, false, false);
  }

  // Epilogue: C/D tile layout -> element j maps to (m = j + 8*(lane>>4), n = lane&15)
  const int   n     = lane & 15;
  const int   mbase = (lane >> 4) << 3;
  const int   s     = col0 + n;
  const float br_b  = bih[s]          + bhh[s];
  const float bz_b  = bih[SS + s]     + bhh[SS + s];
  const float bn_i  = bih[2 * SS + s];
  const float bn_h  = bhh[2 * SS + s];

#pragma unroll
  for (int j = 0; j < 8; ++j) {
    const int row = r0 + mbase + j;
    const size_t o = (size_t)row * SS + s;
    const float hp = hF[o];
    const float rg = fsigmoid(air[j] + ahr[j] + br_b);
    const float zg = fsigmoid(aiz[j] + ahz[j] + bz_b);
    const float ng = tanhf(ain[j] + bn_i + rg * (ahn[j] + bn_h));
    const float hv = (1.0f - zg) * ng + zg * hp;
    outF[o] = hv;
    outB[o] = (__bf16)hv;
    if (extra) extra[o] = hv;
  }
}

// ---------------------------------------------------------------------------
// Small helper kernels
// ---------------------------------------------------------------------------
__global__ void cvt_pad_bf16(const float* __restrict__ src, __bf16* __restrict__ dst,
                             int rows, int ks, int kd) {
  const int idx = blockIdx.x * blockDim.x + threadIdx.x;
  if (idx >= rows * kd) return;
  const int r = idx / kd, k = idx % kd;
  dst[idx] = (k < ks) ? (__bf16)src[(size_t)r * ks + k] : (__bf16)0.0f;
}

__global__ void zero_f32(float* p, int n) {
  const int i = blockIdx.x * blockDim.x + threadIdx.x;
  if (i < n) p[i] = 0.0f;
}
__global__ void zero_bf16(__bf16* p, int n) {
  const int i = blockIdx.x * blockDim.x + threadIdx.x;
  if (i < n) p[i] = (__bf16)0.0f;
}

__global__ void init_dinp(const float* __restrict__ x, float* __restrict__ dF,
                          __bf16* __restrict__ dB) {
  const int idx = blockIdx.x * blockDim.x + threadIdx.x;   // B*CP threads
  if (idx >= BB * CP) return;
  const int b = idx / CP, c = idx % CP;
  if (c < CC) {
    const float v = x[(size_t)b * TT * CC + (size_t)(TT - 1) * CC + c];
    dF[b * CC + c] = v;
    dB[idx] = (__bf16)v;
  } else {
    dB[idx] = (__bf16)0.0f;
  }
}

// temp[b,s] = bt + sum_w Wt[w] * window[b,w,s]   (window = ring slots (i+1+w)&31)
__global__ void win_temp(const float* __restrict__ ring, const float* __restrict__ Wt,
                         const float* __restrict__ bt, float* __restrict__ temp, int i) {
  const int idx = blockIdx.x * blockDim.x + threadIdx.x;   // B*S threads
  if (idx >= BB * SS) return;
  float acc = bt[0];
#pragma unroll
  for (int w = 0; w < WW; ++w) {
    const int slot = (i + 1 + w) & (WW - 1);
    acc += Wt[w] * ring[(size_t)slot * BB * SS + idx];
  }
  temp[idx] = acc;
}

// out[b,c] = temp[b,:] . Ws[c,:] + bs[c] + dinp[b,c];  write d_out slice + next dinp
__global__ void dec_out(const float* __restrict__ temp, const float* __restrict__ Ws,
                        const float* __restrict__ bs, float* __restrict__ dout,
                        float* __restrict__ dF, __bf16* __restrict__ dB, int i) {
  const int idx = blockIdx.x * blockDim.x + threadIdx.x;   // B*CP threads
  if (idx >= BB * CP) return;
  const int b = idx / CP, c = idx % CP;
  if (c >= CC) { dB[idx] = (__bf16)0.0f; return; }
  const float* t = temp + (size_t)b * SS;
  const float* w = Ws + (size_t)c * SS;
  float acc = bs[c];
#pragma unroll 4
  for (int s = 0; s < SS; ++s) acc += t[s] * w[s];
  const float nf = acc + dF[b * CC + c];
  dout[(size_t)b * TT * CC + (size_t)i * CC + c] = nf;
  dF[b * CC + c] = nf;
  dB[idx] = (__bf16)nf;
}

// ---------------------------------------------------------------------------
// Launch
// ---------------------------------------------------------------------------
extern "C" void kernel_launch(void* const* d_in, const int* in_sizes, int n_in,
                              void* d_out, int out_size, void* d_ws, size_t ws_size,
                              hipStream_t stream) {
  (void)in_sizes; (void)n_in; (void)out_size; (void)ws_size;
  const float* x    = (const float*)d_in[0];
  const float* Wih0 = (const float*)d_in[1];
  const float* Whh0 = (const float*)d_in[2];
  const float* bih0 = (const float*)d_in[3];
  const float* bhh0 = (const float*)d_in[4];
  const float* Wih1 = (const float*)d_in[5];
  const float* Whh1 = (const float*)d_in[6];
  const float* bih1 = (const float*)d_in[7];
  const float* bhh1 = (const float*)d_in[8];
  const float* Wt   = (const float*)d_in[9];
  const float* bt   = (const float*)d_in[10];
  const float* Ws   = (const float*)d_in[11];
  const float* bs   = (const float*)d_in[12];
  float* out = (float*)d_out;

  // Workspace carve (256B aligned)
  char*  w   = (char*)d_ws;
  size_t off = 0;
  auto alloc = [&](size_t bytes) -> char* {
    char* p = w + off;
    off = (off + bytes + 255) & ~(size_t)255;
    return p;
  };
  __bf16* xb    = (__bf16*)alloc((size_t)BB * TT * CP * 2);
  __bf16* Wih0b = (__bf16*)alloc((size_t)3 * SS * CP * 2);
  __bf16* Whh0b = (__bf16*)alloc((size_t)3 * SS * SS * 2);
  __bf16* Wih1b = (__bf16*)alloc((size_t)3 * SS * SS * 2);
  __bf16* Whh1b = (__bf16*)alloc((size_t)3 * SS * SS * 2);
  float*  h0F[2]; __bf16* h0B[2]; float* h1F[2]; __bf16* h1B[2];
  for (int p = 0; p < 2; ++p) {
    h0F[p] = (float*) alloc((size_t)BB * SS * 4);
    h0B[p] = (__bf16*)alloc((size_t)BB * SS * 2);
    h1F[p] = (float*) alloc((size_t)BB * SS * 4);
    h1B[p] = (__bf16*)alloc((size_t)BB * SS * 2);
  }
  float*  ring  = (float*) alloc((size_t)WW * BB * SS * 4);
  float*  temp  = (float*) alloc((size_t)BB * SS * 4);
  float*  dinpF = (float*) alloc((size_t)BB * CC * 4);
  __bf16* dinpB = (__bf16*)alloc((size_t)BB * CP * 2);

  // --- one-time conversions / init (re-run every call: deterministic) ---
  const int thr = 256;
  cvt_pad_bf16<<<(BB * TT * CP + thr - 1) / thr, thr, 0, stream>>>(x, xb, BB * TT, CC, CP);
  cvt_pad_bf16<<<(3 * SS * CP + thr - 1) / thr, thr, 0, stream>>>(Wih0, Wih0b, 3 * SS, CC, CP);
  cvt_pad_bf16<<<(3 * SS * SS + thr - 1) / thr, thr, 0, stream>>>(Whh0, Whh0b, 3 * SS, SS, SS);
  cvt_pad_bf16<<<(3 * SS * SS + thr - 1) / thr, thr, 0, stream>>>(Wih1, Wih1b, 3 * SS, SS, SS);
  cvt_pad_bf16<<<(3 * SS * SS + thr - 1) / thr, thr, 0, stream>>>(Whh1, Whh1b, 3 * SS, SS, SS);
  for (int p = 0; p < 2; ++p) {
    zero_f32 <<<(BB * SS + thr - 1) / thr, thr, 0, stream>>>(h0F[p], BB * SS);
    zero_f32 <<<(BB * SS + thr - 1) / thr, thr, 0, stream>>>(h1F[p], BB * SS);
    zero_bf16<<<(BB * SS + thr - 1) / thr, thr, 0, stream>>>(h0B[p], BB * SS);
    zero_bf16<<<(BB * SS + thr - 1) / thr, thr, 0, stream>>>(h1B[p], BB * SS);
  }
  init_dinp<<<(BB * CP + thr - 1) / thr, thr, 0, stream>>>(x, dinpF, dinpB);

  const dim3 cellGrid(SS / 16);   // 32 blocks, one per hidden column tile
  const dim3 cellBlk(256);        // 8 waves -> 8 batch row tiles (128 rows)

  // --- encoder: t = 0 .. T-2 ---
  int p = 0;
  for (int t = 0; t < TT - 1; ++t) {
    const int q = p ^ 1;
    gru_cell_wmma<<<cellGrid, cellBlk, 0, stream>>>(
        xb + (size_t)t * CP, TT * CP, CP, h0B[p], h0F[p],
        Wih0b, Whh0b, bih0, bhh0, h0F[q], h0B[q], nullptr);
    float* extra = (t >= TT - 1 - WW) ? ring + (size_t)(t - (TT - 1 - WW)) * BB * SS : nullptr;
    gru_cell_wmma<<<cellGrid, cellBlk, 0, stream>>>(
        h0B[q], SS, SS, h1B[p], h1F[p],
        Wih1b, Whh1b, bih1, bhh1, h1F[q], h1B[q], extra);
    p = q;
  }

  // --- decoder: i = 0 .. T-1 ---
  for (int i = 0; i < TT; ++i) {
    const int q = p ^ 1;
    gru_cell_wmma<<<cellGrid, cellBlk, 0, stream>>>(
        dinpB, CP, CP, h0B[p], h0F[p],
        Wih0b, Whh0b, bih0, bhh0, h0F[q], h0B[q], nullptr);
    gru_cell_wmma<<<cellGrid, cellBlk, 0, stream>>>(
        h0B[q], SS, SS, h1B[p], h1F[p],
        Wih1b, Whh1b, bih1, bhh1, h1F[q], h1B[q],
        ring + (size_t)(i & (WW - 1)) * BB * SS);
    win_temp<<<(BB * SS + thr - 1) / thr, thr, 0, stream>>>(ring, Wt, bt, temp, i);
    dec_out <<<(BB * CP + thr - 1) / thr, thr, 0, stream>>>(temp, Ws, bs, out, dinpF, dinpB, i);
    p = q;
  }
}
